// GNNModelWrapper_46291157516799
// MI455X (gfx1250) — matrix operand order repaired and checked by reading, not compiled, not run on hardware
//
#include <hip/hip_runtime.h>

#define HIDF 64
#define INF_ 128
#define NSLOPE 0.2f

typedef __attribute__((ext_vector_type(16))) _Float16 v16h;
typedef __attribute__((ext_vector_type(8)))  float    v8f;

__device__ __forceinline__ float lrelu(float v) { return v > 0.f ? v : NSLOPE * v; }

// float atomic max via signed/unsigned integer atomics (monotone encodings)
__device__ __forceinline__ void atomicMaxF32(float* addr, float val) {
  if (val >= 0.f) atomicMax((int*)addr, __float_as_int(val));
  else            atomicMin((unsigned int*)addr, __float_as_uint(val));
}

// ---------------- WMMA fragment loaders (wave32 layouts, ISA 7.12.2) ----------------

// A fragment: 16x32 f16 tile of row-major f32 matrix A (ld = row stride).
// Branchless row guard: out-of-range rows read row 0 with scale 0 (keeps EXEC full).
__device__ __forceinline__ v16h load_a_f32(const float* __restrict__ A, int ld, int row,
                                           int nrows, int k0, int lane, float scale) {
  v16h a;
  const bool ok = row < nrows;
  const int r   = ok ? row : 0;
  const float s = ok ? scale : 0.f;
  const float* p = A + (size_t)r * ld + k0 + ((lane >> 4) ? 8 : 0);
#pragma unroll
  for (int h = 0; h < 8; ++h) a[h] = (_Float16)(p[h] * s);
#pragma unroll
  for (int h = 0; h < 8; ++h) a[8 + h] = (_Float16)(p[16 + h] * s);
  return a;
}

// B fragment from pre-swizzled weights: fragment (kc, nt, lane) stored as 16
// contiguous halves -> one aligned 32B vector load (2x global_load_b128).
__device__ __forceinline__ v16h load_b_swz(const _Float16* __restrict__ S, int kc, int nt,
                                           int lane) {
  return *(const v16h*)(S + ((((size_t)kc * 4 + nt) * 32 + lane) << 4));
}

#define WMMA_F16(a, b, c) \
  __builtin_amdgcn_wmma_f32_16x16x32_f16(false, (a), false, (b), (short)0, (c), false, false)

// ---------------- utility kernels ----------------

__global__ void k_fill(float* __restrict__ p, float v, long long n) {
  long long t = blockIdx.x * (long long)blockDim.x + threadIdx.x;
  if (t < n) p[t] = v;
}

// f32 K x 64 weight -> f16 fragment-swizzled layout:
//   t = (((kc*4 + nt)*32 + lane)*16 + h)
//   source element: W[(kc*32 + (lane>=16?8:0) + (h&7) + (h>=8?16:0)) * 64 + nt*16 + (lane&15)]
__global__ void k_swz16(const float* __restrict__ W, _Float16* __restrict__ S, int n) {
  int t = blockIdx.x * blockDim.x + threadIdx.x;
  if (t >= n) return;
  const int h    = t & 15;
  const int lane = (t >> 4) & 31;
  const int nt   = (t >> 9) & 3;
  const int kc   = t >> 11;
  const int k    = kc * 32 + ((lane >> 4) ? 8 : 0) + (h & 7) + ((h & 8) << 1);
  const int col  = nt * 16 + (lane & 15);
  S[t] = (_Float16)W[(size_t)k * HIDF + col];
}

// ---------------- degree / norm ----------------

__global__ void k_edge_deg(const int* __restrict__ ei, long long E,
                           float* __restrict__ degG, float* __restrict__ degS) {
  long long e = blockIdx.x * (long long)blockDim.x + threadIdx.x;
  if (e >= E) return;
  int d = ei[E + e];
  atomicAdd(degG + d, 1.f);
  atomicAdd(degS + d, 1.f);
}

__global__ void k_dinv(const float* __restrict__ degG, float* __restrict__ dinv, int N) {
  int i = blockIdx.x * blockDim.x + threadIdx.x;
  if (i < N) dinv[i] = rsqrtf(degG[i]);  // deg >= 1 due to self loop
}

// ---------------- WMMA GEMM: Y[N x 64] = X[N x K] @ Wswz[K x 64] ----------------

template <int KDIM>
__device__ __forceinline__ void gemm_tile(const float* __restrict__ X, int ldx,
                                          const _Float16* __restrict__ S, int row, int N,
                                          int lane, float scale, v8f& c0, v8f& c1, v8f& c2,
                                          v8f& c3) {
#pragma unroll
  for (int kc = 0; kc < KDIM / 32; ++kc) {
    v16h a  = load_a_f32(X, ldx, row, N, kc * 32, lane, scale);
    v16h b0 = load_b_swz(S, kc, 0, lane);
    v16h b1 = load_b_swz(S, kc, 1, lane);
    v16h b2 = load_b_swz(S, kc, 2, lane);
    v16h b3 = load_b_swz(S, kc, 3, lane);
    c0 = WMMA_F16(a, b0, c0); c1 = WMMA_F16(a, b1, c1);
    c2 = WMMA_F16(a, b2, c2); c3 = WMMA_F16(a, b3, c3);
  }
}

__device__ __forceinline__ void store_tile(float* __restrict__ Y, int m0, int N, int lane,
                                           const v8f& c0, const v8f& c1, const v8f& c2,
                                           const v8f& c3) {
  const int rbase = m0 + ((lane >> 4) ? 8 : 0);
  const int col   = lane & 15;
#pragma unroll
  for (int v = 0; v < 8; ++v) {
    int r = rbase + v;
    if (r < N) {
      float* yr = Y + (size_t)r * HIDF + col;
      yr[0] = c0[v]; yr[16] = c1[v]; yr[32] = c2[v]; yr[48] = c3[v];
    }
  }
}

__global__ void k_gemm_gcn(const float* __restrict__ X, const _Float16* __restrict__ S,
                           float* __restrict__ Y, int N) {
  int wave = (int)((blockIdx.x * (long long)blockDim.x + threadIdx.x) >> 5);
  int lane = threadIdx.x & 31;
  int m0 = wave * 16;
  if (m0 >= N) return;  // wave-uniform exit: EXEC all-ones around WMMA
  v8f c0 = {}, c1 = {}, c2 = {}, c3 = {};
  gemm_tile<INF_>(X, INF_, S, m0 + (lane & 15), N, lane, 1.f, c0, c1, c2, c3);
  store_tile(Y, m0, N, lane, c0, c1, c2, c3);
}

__global__ void k_gemm_gat(const float* __restrict__ X, const _Float16* __restrict__ S,
                           float* __restrict__ Y, int N) {
  int wave = (int)((blockIdx.x * (long long)blockDim.x + threadIdx.x) >> 5);
  int lane = threadIdx.x & 31;
  int m0 = wave * 16;
  if (m0 >= N) return;
  v8f c0 = {}, c1 = {}, c2 = {}, c3 = {};
  gemm_tile<HIDF>(X, HIDF, S, m0 + (lane & 15), N, lane, 1.f, c0, c1, c2, c3);
  store_tile(Y, m0, N, lane, c0, c1, c2, c3);
}

// out = (ssum/deg) @ WL + h2 @ WR + bias  (mean scaling fused into A-fragment load)
__global__ void k_gemm_sage(const float* __restrict__ SSUM, const float* __restrict__ H2,
                            const float* __restrict__ degS, const _Float16* __restrict__ SL,
                            const _Float16* __restrict__ SR, const float* __restrict__ bias,
                            float* __restrict__ Y, int N) {
  int wave = (int)((blockIdx.x * (long long)blockDim.x + threadIdx.x) >> 5);
  int lane = threadIdx.x & 31;
  int m0 = wave * 16;
  if (m0 >= N) return;
  v8f c0 = {}, c1 = {}, c2 = {}, c3 = {};
  const int row = m0 + (lane & 15);
  float scale = 1.f / fmaxf(row < N ? degS[row] : 1.f, 1.f);
  gemm_tile<HIDF>(SSUM, HIDF, SL, row, N, lane, scale, c0, c1, c2, c3);  // mean @ WL
  gemm_tile<HIDF>(H2, HIDF, SR, row, N, lane, 1.f, c0, c1, c2, c3);      // + h2 @ WR
  const int rbase = m0 + ((lane >> 4) ? 8 : 0);
  const int col   = lane & 15;
#pragma unroll
  for (int v = 0; v < 8; ++v) {
    int r = rbase + v;
    if (r < N) {
      float* yr = Y + (size_t)r * HIDF + col;
      yr[0]  = c0[v] + bias[col];
      yr[16] = c1[v] + bias[16 + col];
      yr[32] = c2[v] + bias[32 + col];
      yr[48] = c3[v] + bias[48 + col];
    }
  }
}

// ---------------- GCN aggregation ----------------

// 16 lanes per edge, 4 feats/lane: acc[dst] += norm * h0[src]
__global__ void k_gcn_agg(const int* __restrict__ ei, long long E,
                          const float* __restrict__ H0, const float* __restrict__ dinv,
                          float* __restrict__ acc) {
  long long t = blockIdx.x * (long long)blockDim.x + threadIdx.x;
  long long e = t >> 4;
  if (e >= E) return;
  const int f = (int)(t & 15) * 4;
  const int s = ei[e], d = ei[E + e];
  const float nrm = dinv[s] * dinv[d];
  const float4 hv = *(const float4*)(H0 + (size_t)s * HIDF + f);
  float* a = acc + (size_t)d * HIDF + f;
  atomicAdd(a + 0, nrm * hv.x);
  atomicAdd(a + 1, nrm * hv.y);
  atomicAdd(a + 2, nrm * hv.z);
  atomicAdd(a + 3, nrm * hv.w);
}

// h1 = relu(acc + dinv^2 * h0 + b)   (in place over h0)
__global__ void k_gcn_fin(const float* __restrict__ acc, float* __restrict__ h,
                          const float* __restrict__ dinv, const float* __restrict__ bias,
                          int N) {
  long long t = blockIdx.x * (long long)blockDim.x + threadIdx.x;
  if (t >= (long long)N * HIDF) return;
  int i = (int)(t >> 6), j = (int)(t & 63);
  float di = dinv[i];
  float v = acc[t] + di * di * h[t] + bias[j];
  h[t] = fmaxf(v, 0.f);
}

// ---------------- GAT ----------------

__global__ void k_att(const float* __restrict__ HG, const float* __restrict__ asw,
                      const float* __restrict__ adw, float* __restrict__ a_s,
                      float* __restrict__ a_d, float* __restrict__ mx, int N) {
  int i = blockIdx.x * blockDim.x + threadIdx.x;
  if (i >= N) return;
  const float* hr = HG + (size_t)i * HIDF;
  float as = 0.f, ad = 0.f;
#pragma unroll
  for (int j = 0; j < HIDF; ++j) {
    float hv = hr[j];
    as = fmaf(hv, asw[j], as);
    ad = fmaf(hv, adw[j], ad);
  }
  a_s[i] = as;
  a_d[i] = ad;
  mx[i] = lrelu(as + ad);  // self-loop edge initializes the segment max
}

__global__ void k_gat_max(const int* __restrict__ ei, long long E,
                          const float* __restrict__ a_s, const float* __restrict__ a_d,
                          float* __restrict__ mx) {
  long long e = blockIdx.x * (long long)blockDim.x + threadIdx.x;
  if (e >= E) return;
  int s = ei[e], d = ei[E + e];
  atomicMaxF32(mx + d, lrelu(a_s[s] + a_d[d]));
}

// p = exp(e - m[dst]);  s[dst] += p;  num[dst] += p * hg[src]
__global__ void k_gat_acc(const int* __restrict__ ei, long long E,
                          const float* __restrict__ HG, const float* __restrict__ a_s,
                          const float* __restrict__ a_d, const float* __restrict__ mx,
                          float* __restrict__ ssc, float* __restrict__ num) {
  long long t = blockIdx.x * (long long)blockDim.x + threadIdx.x;
  long long e = t >> 4;
  if (e >= E) return;
  const int f = (int)(t & 15) * 4;
  const int s = ei[e], d = ei[E + e];
  const float p = __expf(lrelu(a_s[s] + a_d[d]) - mx[d]);
  if ((t & 15) == 0) atomicAdd(ssc + d, p);
  const float4 hv = *(const float4*)(HG + (size_t)s * HIDF + f);
  float* a = num + (size_t)d * HIDF + f;
  atomicAdd(a + 0, p * hv.x);
  atomicAdd(a + 1, p * hv.y);
  atomicAdd(a + 2, p * hv.z);
  atomicAdd(a + 3, p * hv.w);
}

// h2 = relu((num + p_self*hg) / (s + p_self) + b)
__global__ void k_gat_fin(const float* __restrict__ num, const float* __restrict__ HG,
                          const float* __restrict__ a_s, const float* __restrict__ a_d,
                          const float* __restrict__ mx, const float* __restrict__ ssc,
                          const float* __restrict__ bias, float* __restrict__ h2, int N) {
  long long t = blockIdx.x * (long long)blockDim.x + threadIdx.x;
  if (t >= (long long)N * HIDF) return;
  int i = (int)(t >> 6), j = (int)(t & 63);
  float ps = __expf(lrelu(a_s[i] + a_d[i]) - mx[i]);
  float denom = ssc[i] + ps;
  float v = (num[t] + ps * HG[t]) / denom + bias[j];
  h2[t] = fmaxf(v, 0.f);
}

// ---------------- SAGE aggregation ----------------

__global__ void k_sage_agg(const int* __restrict__ ei, long long E,
                           const float* __restrict__ H2, float* __restrict__ ssum) {
  long long t = blockIdx.x * (long long)blockDim.x + threadIdx.x;
  long long e = t >> 4;
  if (e >= E) return;
  const int f = (int)(t & 15) * 4;
  const int s = ei[e], d = ei[E + e];
  const float4 hv = *(const float4*)(H2 + (size_t)s * HIDF + f);
  float* a = ssum + (size_t)d * HIDF + f;
  atomicAdd(a + 0, hv.x);
  atomicAdd(a + 1, hv.y);
  atomicAdd(a + 2, hv.z);
  atomicAdd(a + 3, hv.w);
}

// ---------------- launch ----------------

extern "C" void kernel_launch(void* const* d_in, const int* in_sizes, int n_in,
                              void* d_out, int out_size, void* d_ws, size_t ws_size,
                              hipStream_t stream) {
  const float* x       = (const float*)d_in[0];
  const int*   ei      = (const int*)d_in[1];
  const float* gcn_w   = (const float*)d_in[2];
  const float* gcn_b   = (const float*)d_in[3];
  const float* gat_w   = (const float*)d_in[4];
  const float* att_src = (const float*)d_in[5];
  const float* att_dst = (const float*)d_in[6];
  const float* gat_b   = (const float*)d_in[7];
  const float* sage_wl = (const float*)d_in[8];
  const float* sage_wr = (const float*)d_in[9];
  const float* sage_b  = (const float*)d_in[10];

  const int       N  = in_sizes[0] / INF_;
  const long long E  = (long long)in_sizes[1] / 2;
  const long long NF = (long long)N * HIDF;

  // workspace layout (floats): B0(h0/h1/h2), B1(acc/num/ssum), B2(hg), node arrays, f16 weights
  float* ws    = (float*)d_ws;
  float* B0    = ws;
  float* B1    = B0 + NF;
  float* B2    = B1 + NF;
  float* deg_g = B2 + NF;
  float* dinv  = deg_g + N;
  float* deg_s = dinv + N;
  float* a_s   = deg_s + N;
  float* a_d   = a_s + N;
  float* mx    = a_d + N;
  float* ssc   = mx + N;
  // 32B-align the f16 fragment buffers for the v16h vector loads
  uintptr_t wraw = (uintptr_t)(ssc + N);
  _Float16* gcnw16 = (_Float16*)((wraw + 31) & ~(uintptr_t)31);
  _Float16* gatw16 = gcnw16 + INF_ * HIDF;
  _Float16* wl16   = gatw16 + HIDF * HIDF;
  _Float16* wr16   = wl16 + HIDF * HIDF;

  const int B = 256;
  auto cdiv = [](long long a, long long b) -> unsigned { return (unsigned)((a + b - 1) / b); };
  const unsigned gemm_blk = cdiv((long long)cdiv(N, 16) * 32, B);

  // weight conversion f32 -> f16, swizzled to WMMA B-fragment order
  k_swz16<<<cdiv(INF_ * HIDF, B), B, 0, stream>>>(gcn_w, gcnw16, INF_ * HIDF);
  k_swz16<<<cdiv(HIDF * HIDF, B), B, 0, stream>>>(gat_w, gatw16, HIDF * HIDF);
  k_swz16<<<cdiv(HIDF * HIDF, B), B, 0, stream>>>(sage_wl, wl16, HIDF * HIDF);
  k_swz16<<<cdiv(HIDF * HIDF, B), B, 0, stream>>>(sage_wr, wr16, HIDF * HIDF);

  // degrees (GCN deg includes self loop -> init 1)
  k_fill<<<cdiv(N, B), B, 0, stream>>>(deg_g, 1.f, N);
  k_fill<<<cdiv(N, B), B, 0, stream>>>(deg_s, 0.f, N);
  k_edge_deg<<<cdiv(E, B), B, 0, stream>>>(ei, E, deg_g, deg_s);
  k_dinv<<<cdiv(N, B), B, 0, stream>>>(deg_g, dinv, N);

  // ---- GCN ----
  k_gemm_gcn<<<gemm_blk, B, 0, stream>>>(x, gcnw16, B0, N);            // h0 = x @ W
  k_fill<<<cdiv(NF, B), B, 0, stream>>>(B1, 0.f, NF);
  k_gcn_agg<<<cdiv(E * 16, B), B, 0, stream>>>(ei, E, B0, dinv, B1);   // edge scatter
  k_gcn_fin<<<cdiv(NF, B), B, 0, stream>>>(B1, B0, dinv, gcn_b, N);    // h1 (in place)

  // ---- GAT ----
  k_gemm_gat<<<gemm_blk, B, 0, stream>>>(B0, gatw16, B2, N);           // hg = h1 @ W
  k_att<<<cdiv(N, B), B, 0, stream>>>(B2, att_src, att_dst, a_s, a_d, mx, N);
  k_gat_max<<<cdiv(E, B), B, 0, stream>>>(ei, E, a_s, a_d, mx);
  k_fill<<<cdiv(NF, B), B, 0, stream>>>(B1, 0.f, NF);
  k_fill<<<cdiv(N, B), B, 0, stream>>>(ssc, 0.f, N);
  k_gat_acc<<<cdiv(E * 16, B), B, 0, stream>>>(ei, E, B2, a_s, a_d, mx, ssc, B1);
  k_gat_fin<<<cdiv(NF, B), B, 0, stream>>>(B1, B2, a_s, a_d, mx, ssc, gat_b, B0, N); // h2 -> B0

  // ---- SAGE ----
  k_fill<<<cdiv(NF, B), B, 0, stream>>>(B1, 0.f, NF);
  k_sage_agg<<<cdiv(E * 16, B), B, 0, stream>>>(ei, E, B0, B1);
  k_gemm_sage<<<gemm_blk, B, 0, stream>>>(B1, B0, deg_s, wl16, wr16, sage_b,
                                          (float*)d_out, N);
}
